// STGNNSpoofingDetector_35390530519601
// MI455X (gfx1250) — matrix operand order repaired and checked by reading, compiled, not run on hardware
//
#include <hip/hip_runtime.h>
#include <hip/hip_bf16.h>

// ---------------------------------------------------------------------------
// Types
// ---------------------------------------------------------------------------
typedef __bf16 bf16_t;
typedef __attribute__((ext_vector_type(16))) __bf16 v16bf;
typedef __attribute__((ext_vector_type(8)))  float  v8f;

union FragAB { uint4 u[2]; v16bf v; };
union Frag8  { v8f v; float f[8]; };

#define T_STEPS 8
#define N_NODES 8192
#define N_EDGES 65536
#define N_GRAPH 256

__device__ __forceinline__ bf16_t f2bf(float f) {
  unsigned u = __float_as_uint(f);
  unsigned r = u + 0x7fffu + ((u >> 16) & 1u);   // RNE
  unsigned short s = (unsigned short)(r >> 16);
  bf16_t b; __builtin_memcpy(&b, &s, 2); return b;
}
__device__ __forceinline__ float bf2f(bf16_t b) {
  unsigned short s; __builtin_memcpy(&s, &b, 2);
  return __uint_as_float(((unsigned)s) << 16);
}
__device__ __forceinline__ float bfLo(unsigned w) { return __uint_as_float(w << 16); }
__device__ __forceinline__ float bfHi(unsigned w) { return __uint_as_float(w & 0xffff0000u); }
__device__ __forceinline__ float sigmf(float x) { return 1.f / (1.f + __expf(-x)); }
__device__ __forceinline__ float eluf(float x)  { return x > 0.f ? x : (__expf(x) - 1.f); }
__device__ __forceinline__ int   keyEnc(float f) { int i = __float_as_int(f); return i >= 0 ? i : (i ^ 0x7fffffff); }
__device__ __forceinline__ float keyDec(int k) {
  if (k == (int)0x80000000) return 0.f;            // empty segment -> 0 (matches isfinite fixup)
  return __int_as_float(k >= 0 ? k : (k ^ 0x7fffffff));
}

// ---------------------------------------------------------------------------
// WMMA bf16 GEMM: D[M,N] = A[M,K] * Bt[N,K]^T (+bias[N])
//   A row-major bf16 (lda=K), Bt row-major bf16 (N x K, weight kept in [out,in]),
//   D row-major (f32 or bf16). Block tile 128x64, 8 waves, wave = 16x64 strip.
//   ZPAD=true: K==16, zero-fill LDS past K (edge-feature GEMM only).
//   Register double-buffer: tile k+1 is in flight (global_load_b128) while
//   tile k is consumed from LDS by 4 chained v_wmma_f32_16x16x32_bf16.
// ---------------------------------------------------------------------------
__device__ __forceinline__ uint4 ldTile(const bf16_t* P, int ld, int row0,
                                        int seg, int k0, int K, bool zpad) {
  int r = seg >> 2, q = seg & 3, kk = k0 + q * 8;
  if (zpad && kk >= K) { uint4 z; z.x = z.y = z.z = z.w = 0u; return z; }
  return *(const uint4*)(P + (size_t)(row0 + r) * ld + kk);
}

template<bool OUT_BF16, bool ZPAD>
__global__ __launch_bounds__(256) void gemm_wmma(
    const bf16_t* __restrict__ A, const bf16_t* __restrict__ Bt,
    const float* __restrict__ bias, void* __restrict__ D,
    int M, int N, int K) {
  __shared__ bf16_t sA[128 * 32];
  __shared__ bf16_t sB[64 * 32];
  const int tid  = threadIdx.x;
  const int lane = tid & 31, wave = tid >> 5;
  const int m0 = blockIdx.y * 128, n0 = blockIdx.x * 64;
  const int h = lane >> 4, ln = lane & 15;

  Frag8 acc[4];
  #pragma unroll
  for (int nt = 0; nt < 4; ++nt)
    #pragma unroll
    for (int r = 0; r < 8; ++r) acc[nt].f[r] = 0.f;

  // prefetch tile 0 into registers
  uint4 rA0 = ldTile(A,  K, m0, tid,       0, K, ZPAD);
  uint4 rA1 = ldTile(A,  K, m0, tid + 256, 0, K, ZPAD);
  uint4 rB  = ldTile(Bt, K, n0, tid,       0, K, ZPAD);

  for (int k0 = 0; k0 < K; k0 += 32) {
    // commit staged registers to LDS
    { int r = tid >> 2, q = tid & 3;        *(uint4*)(sA + r * 32 + q * 8) = rA0; }
    { int seg = tid + 256; int r = seg >> 2, q = seg & 3;
                                            *(uint4*)(sA + r * 32 + q * 8) = rA1; }
    { int r = tid >> 2, q = tid & 3;        *(uint4*)(sB + r * 32 + q * 8) = rB;  }
    __syncthreads();

    // kick off next tile's global loads (overlap with WMMA below)
    if (!ZPAD && k0 + 32 < K) {
      rA0 = ldTile(A,  K, m0, tid,       k0 + 32, K, false);
      rA1 = ldTile(A,  K, m0, tid + 256, k0 + 32, K, false);
      rB  = ldTile(Bt, K, n0, tid,       k0 + 32, K, false);
      if (k0 + 64 < K) {  // speculative prefetch of the tile after next
        __builtin_prefetch(A  + (size_t)(m0 + (tid >> 2)) * K + k0 + 64, 0, 0);
        __builtin_prefetch(Bt + (size_t)(n0 + (tid >> 2)) * K + k0 + 64, 0, 0);
      }
    }

    // A fragment: lane holds M=ln, K chunks [h*8, h*8+8) and [16+h*8, 16+h*8+8)
    FragAB fa;
    const int mrow = wave * 16 + ln;
    fa.u[0] = *(const uint4*)(sA + mrow * 32 + h * 8);
    fa.u[1] = *(const uint4*)(sA + mrow * 32 + 16 + h * 8);

    #pragma unroll
    for (int nt = 0; nt < 4; ++nt) {
      FragAB fb;
      const int nrow = nt * 16 + ln;
      fb.u[0] = *(const uint4*)(sB + nrow * 32 + h * 8);
      fb.u[1] = *(const uint4*)(sB + nrow * 32 + 16 + h * 8);
      acc[nt].v = __builtin_amdgcn_wmma_f32_16x16x32_bf16(
          false, fa.v, false, fb.v, (short)0, acc[nt].v, false, false);
    }
    __syncthreads();
  }

  // epilogue: lane holds D[m0+wave*16 + h*8 + r][n0 + nt*16 + ln]
  const int gm = m0 + wave * 16 + h * 8;
  #pragma unroll
  for (int nt = 0; nt < 4; ++nt) {
    int gn = n0 + nt * 16 + ln;
    float bv = bias ? bias[gn] : 0.f;
    #pragma unroll
    for (int r = 0; r < 8; ++r) {
      float val = acc[nt].f[r] + bv;
      size_t idx = (size_t)(gm + r) * N + gn;
      if (OUT_BF16) ((bf16_t*)D)[idx] = f2bf(val);
      else          ((float*)D)[idx]  = val;
    }
  }
}

// ---------------------------------------------------------------------------
// Param packing: fuse Wq|Wk|Wv|Ws per layer into one bf16 weight (row = out),
// fuse edge weights (We1|We2, K=16), fuse biases, precompute BN scale/shift.
// ---------------------------------------------------------------------------
struct Ptrs { const float* p[52]; };

__device__ __forceinline__ void bnPre(int i, const float* g, const float* b,
                                      const float* rm, const float* rv,
                                      float* s, float* sh) {
  float sc = g[i] * rsqrtf(rv[i] + 1e-5f);
  s[i] = sc; sh[i] = b[i] - rm[i] * sc;
}

__global__ __launch_bounds__(256) void pack_params(
    Ptrs P, bf16_t* Wf1, bf16_t* Wef, bf16_t* Wf2, float* bias1, float* bias2,
    float* bnNS, float* bnNB, float* bnES, float* bnEB,
    float* bn1S, float* bn1B, float* bn2S, float* bn2B) {
  int i = blockIdx.x * blockDim.x + threadIdx.x;
  if (i < 32768) {                       // Wf1: 1024 x 32  (q,k,v,s of conv1)
    int row = i >> 5, col = i & 31;
    const float* W[4] = {P.p[20], P.p[22], P.p[24], P.p[27]};
    Wf1[i] = f2bf(W[row >> 8][(row & 255) * 32 + col]);
    return;
  } i -= 32768;
  if (i < 5120) {                        // Wef: 320 x 16   (We1 | We2)
    int row = i >> 4, col = i & 15;
    Wef[i] = f2bf(row < 256 ? P.p[26][row * 16 + col]
                            : P.p[36][(row - 256) * 16 + col]);
    return;
  } i -= 5120;
  if (i < 65536) {                       // Wf2: 256 x 256  (q,k,v,s of conv2)
    int row = i >> 8, col = i & 255;
    const float* W[4] = {P.p[30], P.p[32], P.p[34], P.p[37]};
    Wf2[i] = f2bf(W[row >> 6][(row & 63) * 256 + col]);
    return;
  } i -= 65536;
  if (i < 1024) {                        // bias1
    const float* bb[4] = {P.p[21], P.p[23], P.p[25], P.p[28]};
    bias1[i] = bb[i >> 8][i & 255]; return;
  } i -= 1024;
  if (i < 256) {                         // bias2
    const float* bb[4] = {P.p[31], P.p[33], P.p[35], P.p[38]};
    bias2[i] = bb[i >> 6][i & 63]; return;
  } i -= 256;
  if (i < 32)  { bnPre(i, P.p[4],  P.p[5],  P.p[6],  P.p[7],  bnNS, bnNB); return; } i -= 32;
  if (i < 16)  { bnPre(i, P.p[8],  P.p[9],  P.p[10], P.p[11], bnES, bnEB); return; } i -= 16;
  if (i < 256) { bnPre(i, P.p[12], P.p[13], P.p[14], P.p[15], bn1S, bn1B); return; } i -= 256;
  if (i < 64)  { bnPre(i, P.p[16], P.p[17], P.p[18], P.p[19], bn2S, bn2B); }
}

// ---------------------------------------------------------------------------
// BatchNorm (eval) + convert to bf16
// ---------------------------------------------------------------------------
__global__ __launch_bounds__(256) void bn_to_bf16(
    const float* __restrict__ x, const float* __restrict__ s,
    const float* __restrict__ b, bf16_t* __restrict__ out, int n, int Cmask) {
  int i = blockIdx.x * blockDim.x + threadIdx.x;
  if (i >= n) return;
  int c = i & Cmask;                     // C is a power of 2 (32 / 16)
  out[i] = f2bf(x[i] * s[c] + b[c]);
}

__global__ __launch_bounds__(256) void fill_f32(float* p, int n, float v) {
  int i = blockIdx.x * blockDim.x + threadIdx.x; if (i < n) p[i] = v;
}
__global__ __launch_bounds__(256) void fill_i32(int* p, int n, int v) {
  int i = blockIdx.x * blockDim.x + threadIdx.x; if (i < n) p[i] = v;
}

// ---------------------------------------------------------------------------
// Attention: 3 edge passes (logit+max, exp+denom, aggregate). OUT dim = 64.
// qkvs row = [q | k | v | s], ef row stride 320 (conv1 at 0, conv2 at 256).
// All per-edge reads are float4 / uint4 (8 bf16) vector loads.
// ---------------------------------------------------------------------------
template<int HL>
__global__ __launch_bounds__(256) void attn_logits(
    const float* __restrict__ qkvs, const bf16_t* __restrict__ ef, int efOff,
    const int* __restrict__ src, const int* __restrict__ dst,
    float* __restrict__ logits, int* __restrict__ mbuf) {
  constexpr int HO = HL * 64, RS = 4 * HO;
  int idx = blockIdx.x * blockDim.x + threadIdx.x;
  if (idx >= N_EDGES * HL) return;
  int e = idx / HL, hh = idx % HL;
  int s = src[e], d = dst[e];
  const float4* q  = (const float4*)(qkvs + (size_t)d * RS + hh * 64);
  const float4* k  = (const float4*)(qkvs + (size_t)s * RS + HO + hh * 64);
  const bf16_t* ee = ef + (size_t)e * 320 + efOff + hh * 64;
  float acc = 0.f;
  #pragma unroll
  for (int i = 0; i < 8; ++i) {          // 8 x (8 elements)
    float4 q0 = q[2 * i], q1 = q[2 * i + 1];
    float4 k0 = k[2 * i], k1 = k[2 * i + 1];
    uint4  eu = *(const uint4*)(ee + i * 8);
    acc += q0.x * (k0.x + bfLo(eu.x)) + q0.y * (k0.y + bfHi(eu.x))
         + q0.z * (k0.z + bfLo(eu.y)) + q0.w * (k0.w + bfHi(eu.y))
         + q1.x * (k1.x + bfLo(eu.z)) + q1.y * (k1.y + bfHi(eu.z))
         + q1.z * (k1.z + bfLo(eu.w)) + q1.w * (k1.w + bfHi(eu.w));
  }
  float logit = acc * 0.125f;            // 1/sqrt(64)
  logits[idx] = logit;
  atomicMax(&mbuf[d * HL + hh], keyEnc(logit));
}

template<int HL>
__global__ __launch_bounds__(256) void attn_alpha(
    float* __restrict__ logits, const int* __restrict__ dst,
    const int* __restrict__ mbuf, float* __restrict__ denom) {
  int idx = blockIdx.x * blockDim.x + threadIdx.x;
  if (idx >= N_EDGES * HL) return;
  int e = idx / HL, hh = idx % HL;
  int d = dst[e];
  float a = __expf(logits[idx] - keyDec(mbuf[d * HL + hh]));
  logits[idx] = a;                       // reuse buffer for unnormalized alpha
  atomicAdd(&denom[d * HL + hh], a);
}

template<int HL>
__global__ __launch_bounds__(256) void attn_aggregate(
    const float* __restrict__ qkvs, const bf16_t* __restrict__ ef, int efOff,
    const int* __restrict__ src, const int* __restrict__ dst,
    const float* __restrict__ alpha, float* __restrict__ agg) {
  constexpr int HO = HL * 64, RS = 4 * HO;
  int idx = blockIdx.x * blockDim.x + threadIdx.x;
  if (idx >= N_EDGES * HL) return;
  int e = idx / HL, hh = idx % HL;
  int s = src[e], d = dst[e];
  float a = alpha[idx];
  const float4* v  = (const float4*)(qkvs + (size_t)s * RS + 2 * HO + hh * 64);
  const bf16_t* ee = ef + (size_t)e * 320 + efOff + hh * 64;
  float* out = agg + (size_t)d * HO + hh * 64;
  #pragma unroll
  for (int i = 0; i < 8; ++i) {
    float4 v0 = v[2 * i], v1 = v[2 * i + 1];
    uint4  eu = *(const uint4*)(ee + i * 8);
    atomicAdd(&out[i * 8 + 0], (v0.x + bfLo(eu.x)) * a);
    atomicAdd(&out[i * 8 + 1], (v0.y + bfHi(eu.x)) * a);
    atomicAdd(&out[i * 8 + 2], (v0.z + bfLo(eu.y)) * a);
    atomicAdd(&out[i * 8 + 3], (v0.w + bfHi(eu.y)) * a);
    atomicAdd(&out[i * 8 + 4], (v1.x + bfLo(eu.z)) * a);
    atomicAdd(&out[i * 8 + 5], (v1.y + bfHi(eu.z)) * a);
    atomicAdd(&out[i * 8 + 6], (v1.z + bfLo(eu.w)) * a);
    atomicAdd(&out[i * 8 + 7], (v1.w + bfHi(eu.w)) * a);
  }
}

// ---------------------------------------------------------------------------
// Finalize conv1: normalize agg, beta-gated skip, BN1+ELU -> bf16 [N,256]
// Wave-per-node, wave32 shuffle reduction for the beta dot product.
// ---------------------------------------------------------------------------
__global__ __launch_bounds__(256) void finalize1(
    const float* __restrict__ agg, const float* __restrict__ den,
    const float* __restrict__ qkvs, const float* __restrict__ Wb,
    const float* __restrict__ bnS, const float* __restrict__ bnB,
    bf16_t* __restrict__ x1b) {
  int wave = threadIdx.x >> 5, lane = threadIdx.x & 31;
  int n = blockIdx.x * 8 + wave;
  if (n >= N_NODES) return;
  float av[8], rv_[8], part = 0.f;
  #pragma unroll
  for (int ii = 0; ii < 8; ++ii) {
    int j = lane + ii * 32;
    float a = agg[(size_t)n * 256 + j] / (den[n * 4 + (j >> 6)] + 1e-16f);
    float r = qkvs[(size_t)n * 1024 + 768 + j];
    av[ii] = a; rv_[ii] = r;
    part += Wb[j] * a + Wb[256 + j] * r + Wb[512 + j] * (a - r);
  }
  #pragma unroll
  for (int o = 16; o > 0; o >>= 1) part += __shfl_xor(part, o, 32);
  float beta = sigmf(part);
  #pragma unroll
  for (int ii = 0; ii < 8; ++ii) {
    int j = lane + ii * 32;
    float x = beta * rv_[ii] + (1.f - beta) * av[ii];
    x1b[(size_t)n * 256 + j] = f2bf(eluf(bnS[j] * x + bnB[j]));
  }
}

// Finalize conv2 (heads=1, 64 dims), BN2+ELU -> f32 [N,64]
__global__ __launch_bounds__(256) void finalize2(
    const float* __restrict__ agg, const float* __restrict__ den,
    const float* __restrict__ qkvs, const float* __restrict__ Wb,
    const float* __restrict__ bnS, const float* __restrict__ bnB,
    float* __restrict__ x2) {
  int wave = threadIdx.x >> 5, lane = threadIdx.x & 31;
  int n = blockIdx.x * 8 + wave;
  if (n >= N_NODES) return;
  float av[2], rv_[2], part = 0.f;
  #pragma unroll
  for (int ii = 0; ii < 2; ++ii) {
    int j = lane + ii * 32;
    float a = agg[(size_t)n * 64 + j] / (den[n] + 1e-16f);
    float r = qkvs[(size_t)n * 256 + 192 + j];
    av[ii] = a; rv_[ii] = r;
    part += Wb[j] * a + Wb[64 + j] * r + Wb[128 + j] * (a - r);
  }
  #pragma unroll
  for (int o = 16; o > 0; o >>= 1) part += __shfl_xor(part, o, 32);
  float beta = sigmf(part);
  #pragma unroll
  for (int ii = 0; ii < 2; ++ii) {
    int j = lane + ii * 32;
    float x = beta * rv_[ii] + (1.f - beta) * av[ii];
    x2[(size_t)n * 64 + j] = eluf(bnS[j] * x + bnB[j]);
  }
}

// ---------------------------------------------------------------------------
// Per-graph mean/max pool (batch = repeat(arange(256), 32), contiguous blocks)
// ---------------------------------------------------------------------------
__global__ __launch_bounds__(64) void pool_kernel(
    const float* __restrict__ x2, float* __restrict__ embs, int t) {
  int g = blockIdx.x, d = threadIdx.x;
  float sum = 0.f, mx = -3.4e38f;
  #pragma unroll 4
  for (int i = 0; i < 32; ++i) {
    float v = x2[(size_t)(g * 32 + i) * 64 + d];
    sum += v; mx = fmaxf(mx, v);
  }
  float* row = embs + ((size_t)t * N_GRAPH + g) * 128;
  row[d] = sum * (1.f / 32.f);
  row[64 + d] = mx;
}

// ---------------------------------------------------------------------------
// LSTM: one block per graph, thread j owns unit j; h in LDS, c in reg.
// Gate order i,f,g,o. Used for fwd full scan and bwd single step.
// ---------------------------------------------------------------------------
__global__ __launch_bounds__(128) void lstm_kernel(
    const float* __restrict__ embs, const float* __restrict__ Wih,
    const float* __restrict__ Whh, const float* __restrict__ bih,
    const float* __restrict__ bhh, float* __restrict__ temporal,
    int tStart, int tStep, int nSteps, int outOff) {
  __shared__ __align__(16) float hs[128];
  __shared__ __align__(16) float xs[128];
  int b = blockIdx.x, j = threadIdx.x;
  float c = 0.f; hs[j] = 0.f;
  __syncthreads();
  for (int s = 0; s < nSteps; ++s) {
    int t = tStart + s * tStep;
    xs[j] = embs[((size_t)t * N_GRAPH + b) * 128 + j];
    __syncthreads();
    float g[4];
    #pragma unroll
    for (int gt = 0; gt < 4; ++gt) {
      int row = gt * 128 + j;
      float acc = bih[row] + bhh[row];
      const float4* wi = (const float4*)(Wih + (size_t)row * 128);
      const float4* wh = (const float4*)(Whh + (size_t)row * 128);
      const float4* x4 = (const float4*)xs;
      const float4* h4 = (const float4*)hs;
      for (int d = 0; d < 32; ++d) {
        float4 a = wi[d], xv = x4[d], w = wh[d], hv = h4[d];
        acc += a.x * xv.x + a.y * xv.y + a.z * xv.z + a.w * xv.w
             + w.x * hv.x + w.y * hv.y + w.z * hv.z + w.w * hv.w;
      }
      g[gt] = acc;
    }
    c = sigmf(g[1]) * c + sigmf(g[0]) * tanhf(g[2]);
    float hn = sigmf(g[3]) * tanhf(c);
    __syncthreads();
    hs[j] = hn;
    __syncthreads();
  }
  temporal[(size_t)b * 256 + outOff + j] = hs[j];
}

// ---------------------------------------------------------------------------
// Classifier: out[B,2] = elu(temporal @ W1^T + b1) @ W2^T + b2
// ---------------------------------------------------------------------------
__global__ __launch_bounds__(64) void clf_kernel(
    const float* __restrict__ temporal, const float* __restrict__ W1,
    const float* __restrict__ b1, const float* __restrict__ W2,
    const float* __restrict__ b2, float* __restrict__ out) {
  __shared__ __align__(16) float h1[64];
  int b = blockIdx.x, d = threadIdx.x;
  const float4* tr = (const float4*)(temporal + (size_t)b * 256);
  const float4* w  = (const float4*)(W1 + (size_t)d * 256);
  float acc = b1[d];
  for (int k = 0; k < 64; ++k) {
    float4 a = w[k], x = tr[k];
    acc += a.x * x.x + a.y * x.y + a.z * x.z + a.w * x.w;
  }
  h1[d] = eluf(acc);
  __syncthreads();
  if (d < 2) {
    const float* w2 = W2 + (size_t)d * 64;
    float o = b2[d];
    for (int k = 0; k < 64; ++k) o += w2[k] * h1[k];
    out[(size_t)b * 2 + d] = o;
  }
}

// ---------------------------------------------------------------------------
// Host launcher
// ---------------------------------------------------------------------------
extern "C" void kernel_launch(void* const* d_in, const int* in_sizes, int n_in,
                              void* d_out, int out_size, void* d_ws, size_t ws_size,
                              hipStream_t stream) {
  (void)in_sizes; (void)out_size; (void)ws_size;
  const float* x_seq = (const float*)d_in[0];
  const float* ea_seq = (const float*)d_in[1];
  const int*   ei = (const int*)d_in[2];
  const int*   src = ei;
  const int*   dst = ei + N_EDGES;

  Ptrs P;
  for (int i = 0; i < 52 && i < n_in; ++i) P.p[i] = (const float*)d_in[i];

  char* base = (char*)d_ws; size_t off = 0;
  auto alloc = [&](size_t bytes) -> void* {
    off = (off + 255) & ~(size_t)255;
    void* p = base + off; off += bytes; return p;
  };

  float* bnNS = (float*)alloc(32 * 4);  float* bnNB = (float*)alloc(32 * 4);
  float* bnES = (float*)alloc(16 * 4);  float* bnEB = (float*)alloc(16 * 4);
  float* bn1S = (float*)alloc(256 * 4); float* bn1B = (float*)alloc(256 * 4);
  float* bn2S = (float*)alloc(64 * 4);  float* bn2B = (float*)alloc(64 * 4);
  float* bias1 = (float*)alloc(1024 * 4);
  float* bias2 = (float*)alloc(256 * 4);
  bf16_t* Wf1 = (bf16_t*)alloc((size_t)1024 * 32 * 2);
  bf16_t* Wef = (bf16_t*)alloc((size_t)320 * 16 * 2);
  bf16_t* Wf2 = (bf16_t*)alloc((size_t)256 * 256 * 2);
  bf16_t* xb  = (bf16_t*)alloc((size_t)N_NODES * 32 * 2);
  bf16_t* eb  = (bf16_t*)alloc((size_t)N_EDGES * 16 * 2);
  float*  qkvs1 = (float*)alloc((size_t)N_NODES * 1024 * 4);
  bf16_t* ef   = (bf16_t*)alloc((size_t)N_EDGES * 320 * 2);
  float*  logits1 = (float*)alloc((size_t)N_EDGES * 4 * 4);
  int*    mbuf1   = (int*)alloc((size_t)N_NODES * 4 * 4);
  float*  den1    = (float*)alloc((size_t)N_NODES * 4 * 4);
  float*  agg1    = (float*)alloc((size_t)N_NODES * 256 * 4);
  bf16_t* x1b     = (bf16_t*)alloc((size_t)N_NODES * 256 * 2);
  float*  qkvs2   = (float*)alloc((size_t)N_NODES * 256 * 4);
  float*  logits2 = (float*)alloc((size_t)N_EDGES * 4);
  int*    mbuf2   = (int*)alloc((size_t)N_NODES * 4);
  float*  den2    = (float*)alloc((size_t)N_NODES * 4);
  float*  agg2    = (float*)alloc((size_t)N_NODES * 64 * 4);
  float*  x2      = (float*)alloc((size_t)N_NODES * 64 * 4);
  float*  embs    = (float*)alloc((size_t)T_STEPS * N_GRAPH * 128 * 4);
  float*  temporal = (float*)alloc((size_t)N_GRAPH * 256 * 4);

  // ---- pack weights / bias / BN once per call -----------------------------
  {
    int work = 32768 + 5120 + 65536 + 1024 + 256 + 32 + 16 + 256 + 64;
    pack_params<<<(work + 255) / 256, 256, 0, stream>>>(
        P, Wf1, Wef, Wf2, bias1, bias2,
        bnNS, bnNB, bnES, bnEB, bn1S, bn1B, bn2S, bn2B);
  }

  for (int t = 0; t < T_STEPS; ++t) {
    const float* x_t  = x_seq  + (size_t)t * N_NODES * 32;
    const float* ea_t = ea_seq + (size_t)t * N_EDGES * 16;

    // BN + bf16 convert
    bn_to_bf16<<<(N_NODES * 32 + 255) / 256, 256, 0, stream>>>(x_t, bnNS, bnNB, xb, N_NODES * 32, 31);
    bn_to_bf16<<<(N_EDGES * 16 + 255) / 256, 256, 0, stream>>>(ea_t, bnES, bnEB, eb, N_EDGES * 16, 15);

    // fused Q|K|V|S projection conv1: [8192,32] x [32,1024]
    gemm_wmma<false, false><<<dim3(1024 / 64, N_NODES / 128), 256, 0, stream>>>(
        xb, Wf1, bias1, qkvs1, N_NODES, 1024, 32);
    // fused edge projection: [65536,16] x [16,320] -> bf16 (K=16, zero-padded)
    gemm_wmma<true, true><<<dim3(320 / 64, N_EDGES / 128), 256, 0, stream>>>(
        eb, Wef, nullptr, ef, N_EDGES, 320, 16);

    // reset accumulators
    fill_f32<<<(N_NODES * 256 + 255) / 256, 256, 0, stream>>>(agg1, N_NODES * 256, 0.f);
    fill_f32<<<(N_NODES * 4 + 255) / 256, 256, 0, stream>>>(den1, N_NODES * 4, 0.f);
    fill_i32<<<(N_NODES * 4 + 255) / 256, 256, 0, stream>>>(mbuf1, N_NODES * 4, (int)0x80000000);
    fill_f32<<<(N_NODES * 64 + 255) / 256, 256, 0, stream>>>(agg2, N_NODES * 64, 0.f);
    fill_f32<<<(N_NODES + 255) / 256, 256, 0, stream>>>(den2, N_NODES, 0.f);
    fill_i32<<<(N_NODES + 255) / 256, 256, 0, stream>>>(mbuf2, N_NODES, (int)0x80000000);

    // conv1 attention (heads=4)
    int w1 = N_EDGES * 4;
    attn_logits<4><<<(w1 + 255) / 256, 256, 0, stream>>>(qkvs1, ef, 0, src, dst, logits1, mbuf1);
    attn_alpha<4><<<(w1 + 255) / 256, 256, 0, stream>>>(logits1, dst, mbuf1, den1);
    attn_aggregate<4><<<(w1 + 255) / 256, 256, 0, stream>>>(qkvs1, ef, 0, src, dst, logits1, agg1);
    finalize1<<<N_NODES / 8, 256, 0, stream>>>(agg1, den1, qkvs1, P.p[29], bn1S, bn1B, x1b);

    // fused projection conv2: [8192,256] x [256,256]
    gemm_wmma<false, false><<<dim3(256 / 64, N_NODES / 128), 256, 0, stream>>>(
        x1b, Wf2, bias2, qkvs2, N_NODES, 256, 256);

    // conv2 attention (heads=1)
    attn_logits<1><<<(N_EDGES + 255) / 256, 256, 0, stream>>>(qkvs2, ef, 256, src, dst, logits2, mbuf2);
    attn_alpha<1><<<(N_EDGES + 255) / 256, 256, 0, stream>>>(logits2, dst, mbuf2, den2);
    attn_aggregate<1><<<(N_EDGES + 255) / 256, 256, 0, stream>>>(qkvs2, ef, 256, src, dst, logits2, agg2);
    finalize2<<<N_NODES / 8, 256, 0, stream>>>(agg2, den2, qkvs2, P.p[39], bn2S, bn2B, x2);

    // per-graph mean/max pool -> embs[t]
    pool_kernel<<<N_GRAPH, 64, 0, stream>>>(x2, embs, t);
  }

  // BiLSTM: full fwd scan (need h at t=T-1); bwd hidden at time T-1 = 1 step
  lstm_kernel<<<N_GRAPH, 128, 0, stream>>>(embs, P.p[40], P.p[41], P.p[42], P.p[43],
                                           temporal, 0, 1, T_STEPS, 0);
  lstm_kernel<<<N_GRAPH, 128, 0, stream>>>(embs, P.p[44], P.p[45], P.p[46], P.p[47],
                                           temporal, T_STEPS - 1, -1, 1, 128);

  // classifier
  clf_kernel<<<N_GRAPH, 64, 0, stream>>>(temporal, P.p[48], P.p[49], P.p[50], P.p[51],
                                         (float*)d_out);
}
